// CG_model_jit_40355512713743
// MI455X (gfx1250) — matrix operand order repaired and checked by reading, not compiled, not run on hardware
//
#include <hip/hip_runtime.h>

typedef __attribute__((ext_vector_type(16))) _Float16 v16h;
typedef __attribute__((ext_vector_type(8)))  float    v8f;
typedef __attribute__((ext_vector_type(2)))  float    v2f;

#define HINV   (1.0f / 3.0f)   // 1/H, H = 3.0
#define NTILES 8               // 128 features / 16
#define KTILES 4               // 128 features / 32 (K of f16 WMMA)
#define WAVES  4               // waves per block
#define TPB    (WAVES * 32)

// Fast silu: x * sigmoid(x) using v_exp_f32 + v_rcp_f32 (no IEEE div sequence).
__device__ __forceinline__ float silu_fast(float x) {
    return x * __builtin_amdgcn_rcpf(1.0f + __expf(-x));
}

// ---------------- setup kernels ----------------

__global__ void k_zero(float* p, int n) {
    int i = blockIdx.x * blockDim.x + threadIdx.x;
    if (i < n) p[i] = 0.0f;
}

// Pre-arrange W1^T (B matrix of h1 = h0 @ W1^T) as f16 in B-fragment order.
// B fragment (16-bit, 32x16): lane L holds column n = nt*16 + (L&15),
// K = kt*32 + 8*(L>>4) + (t&7) + 16*(t>>3) for t=0..15 (matches A table order).
__global__ void k_convert_w1(const float* __restrict__ W1, _Float16* __restrict__ frag) {
    int idx = blockIdx.x * blockDim.x + threadIdx.x;   // 32 frags * 32 lanes * 16
    if (idx >= 32 * 32 * 16) return;
    int t    = idx & 15;
    int lane = (idx >> 4) & 31;
    int fid  = idx >> 9;
    int nt   = fid & 7;
    int kt   = fid >> 3;
    int n = nt * 16 + (lane & 15);
    int k = kt * 32 + ((lane >> 4) << 3) + (t & 7) + ((t >> 3) << 4);
    frag[idx] = (_Float16)W1[n * 128 + k];             // B[k][n] = W1[n][k]
}

// ---------------- main edge kernel ----------------

__launch_bounds__(TPB)
__global__ void k_edges(const float* __restrict__ v,  const float* __restrict__ r_ij,
                        const float* __restrict__ W0, const float* __restrict__ b0,
                        const float* __restrict__ b1, const float* __restrict__ W2,
                        const float* __restrict__ b2, const long long* __restrict__ ei,
                        const _Float16* __restrict__ w1frag,
                        float* __restrict__ sum_i, float* __restrict__ sum_j,
                        float* __restrict__ cnt_i, float* __restrict__ cnt_j,
                        int E, int ntiles)
{
    __shared__ _Float16 sW1[32 * 512];          // 32 KB: W1^T B-fragments
    __shared__ _Float16 sH0[WAVES][16 * 128];   // per-wave h0 staging [edge][feat]
    __shared__ float    sM[WAVES][16];          // per-wave layer-2 results

    // cooperative block load of W1 fragments into LDS (128-bit copies)
    {
        const uint4* src = (const uint4*)w1frag;
        uint4*       dst = (uint4*)sW1;
        for (int i = threadIdx.x; i < (32 * 512 * 2) / 16; i += TPB) dst[i] = src[i];
    }
    __syncthreads();

    const int w    = threadIdx.x >> 5;
    const int lane = threadIdx.x & 31;
    const int l15  = lane & 15;
    const int hi   = lane >> 4;

    // Static per-lane fragments (held in VGPRs for the whole kernel).
    // Layer-0 B fragment (f32 4x16): V0 = {k0 | k2}, V1 = {k1 | k3} by lane half.
    v2f   bw0[NTILES];
    float b0f[NTILES], b1f[NTILES], w2f[NTILES];
#pragma unroll
    for (int nt = 0; nt < NTILES; ++nt) {
        int n = nt * 16 + l15;
        bw0[nt].x = W0[n * 4 + 2 * hi];
        bw0[nt].y = W0[n * 4 + 2 * hi + 1];
        b0f[nt] = b0[n];
        b1f[nt] = b1[n];
        w2f[nt] = W2[n];
    }
    const float b2s = b2[0];

    _Float16* hp = &sH0[w][0];

    for (int tile = blockIdx.x * WAVES + w; tile < ntiles; tile += gridDim.x * WAVES) {
        // lanes L and L+16 both own edge tile*16 + (L&15)
        const int e = tile * 16 + l15;
        const long long i_e = ei[e];
        const long long j_e = ei[E + e];
        const float rx = r_ij[e * 3 + 0], ry = r_ij[e * 3 + 1], rz = r_ij[e * 3 + 2];
        const float r  = sqrtf(rx * rx + ry * ry + rz * rz) * HINV;
        const float vx = v[i_e * 3 + 0] - v[j_e * 3 + 0];
        const float vy = v[i_e * 3 + 1] - v[j_e * 3 + 1];
        const float vz = v[i_e * 3 + 2] - v[j_e * 3 + 2];

        if (!hi) {  // degree counts (once per edge)
            atomicAdd(&cnt_i[(int)i_e], 1.0f);
            atomicAdd(&cnt_j[(int)j_e], 1.0f);
        }

        for (int s = 0; s < 2; ++s) {       // s=0: +v_ij (m_i), s=1: -v_ij (m_j)
            const float sgn = s ? -1.0f : 1.0f;
            // Layer-0 A fragment (f32 16x4): lanes 0-15 hold (K0,K1)=(r, vx),
            // lanes 16-31 hold (K2,K3)=(vy, vz) of the same row.
            v2f ax;
            ax.x = hi ? sgn * vy : r;
            ax.y = hi ? sgn * vz : sgn * vx;

            // ---- layer 0: [16x4] @ [4x128] via 8x v_wmma_f32_16x16x4_f32 ----
#pragma unroll
            for (int nt = 0; nt < NTILES; ++nt) {
                v8f d = {};
                d = __builtin_amdgcn_wmma_f32_16x16x4_f32(
                        false, ax, false, bw0[nt], (short)0, d, false, false);
                // C layout: element q is (row = q + 8*hi, col = nt*16 + l15)
#pragma unroll
                for (int q = 0; q < 8; ++q) {
                    float h = silu_fast(d[q] + b0f[nt]);
                    hp[(q + 8 * hi) * 128 + nt * 16 + l15] = (_Float16)h;
                }
            }
            asm volatile("s_wait_dscnt 0x0" ::: "memory");

            // ---- layer-1 A fragments from LDS (contiguous b128 reads) ----
            // 16-bit A 16x32: lanes 0-15 -> K {0..7,16..23}; lanes 16-31 -> K {8..15,24..31}
            union AF { v16h vv; uint4 u[2]; } afr[KTILES];
#pragma unroll
            for (int kt = 0; kt < KTILES; ++kt) {
                const _Float16* base = &hp[l15 * 128 + kt * 32 + 8 * hi];
                afr[kt].u[0] = *(const uint4*)(base);
                afr[kt].u[1] = *(const uint4*)(base + 16);
            }

            float oacc[8];
#pragma unroll
            for (int q = 0; q < 8; ++q) oacc[q] = 0.0f;

            // ---- layer 1: 8 N-tiles x 4 K-tiles of v_wmma_f32_16x16x32_f16,
            //      fused with bias+silu and the layer-2 dot-product partials ----
#pragma unroll
            for (int nt = 0; nt < NTILES; ++nt) {
                v8f acc = {};
#pragma unroll
                for (int kt = 0; kt < KTILES; ++kt) {
                    union BF { v16h vv; uint4 u[2]; } bfr;
                    const uint4* bp = (const uint4*)&sW1[((kt * 8 + nt) * 32 + lane) * 16];
                    bfr.u[0] = bp[0];
                    bfr.u[1] = bp[1];
                    acc = __builtin_amdgcn_wmma_f32_16x16x32_f16(
                            false, afr[kt].vv, false, bfr.vv, (short)0, acc, false, false);
                }
#pragma unroll
                for (int q = 0; q < 8; ++q) {
                    float h = silu_fast(acc[q] + b1f[nt]);
                    oacc[q] += h * w2f[nt];                      // layer-2 partial
                }
            }

            // finish layer-2 dot: butterfly reduce across each 16-lane half
#pragma unroll
            for (int m = 1; m < 16; m <<= 1) {
#pragma unroll
                for (int q = 0; q < 8; ++q) oacc[q] += __shfl_xor(oacc[q], m, 32);
            }

            if (l15 == 0) {  // lane 0 -> edges 0..7, lane 16 -> edges 8..15
#pragma unroll
                for (int q = 0; q < 8; ++q) sM[w][8 * hi + q] = oacc[q] + b2s;
            }
            asm volatile("s_wait_dscnt 0x0" ::: "memory");

            if (!hi) {  // one atomic scatter-add per edge per sign
                float mval = sM[w][l15];
                int   idx  = s ? (int)j_e : (int)i_e;
                atomicAdd((s ? sum_j : sum_i) + idx, mval);
            }
        }
    }
}

// ---------------- finalize: mean_i + mean_j ----------------

__global__ void k_final(const float* __restrict__ sum_i, const float* __restrict__ sum_j,
                        const float* __restrict__ cnt_i, const float* __restrict__ cnt_j,
                        float* __restrict__ out, int n)
{
    int i = blockIdx.x * blockDim.x + threadIdx.x;
    if (i < n)
        out[i] = sum_i[i] * __builtin_amdgcn_rcpf(fmaxf(cnt_i[i], 1.0f))
               + sum_j[i] * __builtin_amdgcn_rcpf(fmaxf(cnt_j[i], 1.0f));
}

// ---------------- launcher ----------------

extern "C" void kernel_launch(void* const* d_in, const int* in_sizes, int n_in,
                              void* d_out, int out_size, void* d_ws, size_t ws_size,
                              hipStream_t stream)
{
    const float*     v    = (const float*)d_in[0];
    const float*     r_ij = (const float*)d_in[1];
    const float*     W0   = (const float*)d_in[2];
    const float*     b0   = (const float*)d_in[3];
    const float*     W1   = (const float*)d_in[4];
    const float*     b1   = (const float*)d_in[5];
    const float*     W2   = (const float*)d_in[6];
    const float*     b2   = (const float*)d_in[7];
    const long long* ei   = (const long long*)d_in[8];
    float*           out  = (float*)d_out;

    const int N = in_sizes[0] / 3;       // 100000 nodes
    const int E = in_sizes[1] / 3;       // 1.6M edges

    float* sum_i = (float*)d_ws;
    float* sum_j = sum_i + N;
    float* cnt_i = sum_j + N;
    float* cnt_j = cnt_i + N;
    _Float16* w1frag = (_Float16*)(cnt_j + N);   // 16N bytes offset -> 16B aligned

    k_zero<<<(4 * N + 255) / 256, 256, 0, stream>>>(sum_i, 4 * N);
    k_convert_w1<<<(32 * 32 * 16 + 255) / 256, 256, 0, stream>>>(W1, w1frag);

    const int ntiles = E / 16;           // 100000 tiles of 16 edges
    k_edges<<<1024, TPB, 0, stream>>>(v, r_ij, W0, b0, b1, W2, b2, ei, w1frag,
                                      sum_i, sum_j, cnt_i, cnt_j, E, ntiles);

    k_final<<<(N + 255) / 256, 256, 0, stream>>>(sum_i, sum_j, cnt_i, cnt_j, out, N);
}
